// SwinTransformerBlock_11020886082056
// MI455X (gfx1250) — compile-verified, matrix operand order checked by reading, not run on hardware
//
#include <hip/hip_runtime.h>
#include <math.h>

// ---------------- CDNA5 WMMA plumbing (wave32, 16x16x32 bf16 -> f32) -------
typedef __bf16 bf16;
typedef __attribute__((ext_vector_type(16))) __bf16 v16bf;
typedef __attribute__((ext_vector_type(8)))  float  v8f;

// Async global->LDS b128 builtin parameter types (per hipcc diagnostic:
// "int __attribute__((vector_size(16))) __device__ *").
typedef int gv4i __attribute__((vector_size(16)));
typedef __attribute__((address_space(1))) void* as1_void_p;
typedef __attribute__((address_space(3))) void* as3_void_p;
typedef __attribute__((address_space(1))) gv4i* as1_v4i_p;
typedef __attribute__((address_space(3))) gv4i* as3_v4i_p;

#define C_    96
#define NTOK  49
#define NPAD  64
#define HD_   32
#define NH_   3
#define HW_   224
#define SH_   3
#define LTOT  (HW_*HW_)     // 50176 tokens per batch image
#define BB_   8
#define HID_  384

// Padded LDS strides (bank-conflict avoidance; must stay even for pair loads)
#define XLD   104           // 96 -> 104
#define WLD   104
#define QLD   296           // 288 -> 296 (unified qkv panel)
#define VLD   72            // 64 -> 72
#define SLD   65            // f32 score rows
#define PLD   72
#define OLD   104
#define ALD   392           // 384 -> 392

// Load a 16x32 bf16 fragment (A or B operand) from an LDS row-major panel.
// lane%16 selects the operand row; lane/16 selects the k-subgroups per the
// ISA 16-bit layout: k = {8h+0..7} then {16+8h+0..7}, pair-packed per VGPR.
__device__ __forceinline__ v16bf load_frag(const bf16* base, int row, int ld,
                                           int k0, int lane) {
  union { v16bf v; unsigned u[8]; } f;
  const int h = (lane >> 4) & 1;
  const unsigned* p = (const unsigned*)(base + row * ld + k0 + 8 * h);
  const unsigned* q = (const unsigned*)(base + row * ld + k0 + 16 + 8 * h);
  f.u[0] = p[0]; f.u[1] = p[1]; f.u[2] = p[2]; f.u[3] = p[3];
  f.u[4] = q[0]; f.u[5] = q[1]; f.u[6] = q[2]; f.u[7] = q[3];
  return f.v;
}

__device__ __forceinline__ v8f wmma_bf16(v16bf a, v16bf b, v8f c) {
  return __builtin_amdgcn_wmma_f32_16x16x32_bf16(false, a, false, b,
                                                 (short)0, c, false, false);
}

// ===========================================================================
// Kernel A: LN1 + shifted-window gather + QKV + attention + proj -> h (d_out)
// One workgroup per 7x7 window (padded to 64 tokens). 256 threads = 8 waves.
// ===========================================================================
__global__ __launch_bounds__(256) void swin_attn_kernel(
    const float* __restrict__ x,
    const float* __restrict__ ln1g, const float* __restrict__ ln1b,
    const float* __restrict__ wqkv, const float* __restrict__ bqkv,
    const float* __restrict__ wproj, const float* __restrict__ bproj,
    float* __restrict__ hout) {
  __shared__ __align__(16) bf16  sX  [NPAD * XLD];      // LN'd window, bf16
  __shared__ __align__(16) bf16  sWq [3 * C_ * WLD];    // wqkv^T: [n][k]
  __shared__ __align__(16) bf16  sWp [C_ * WLD];        // wproj^T: [n][k]
  __shared__ __align__(16) bf16  sQKV[NPAD * QLD];      // [token][288] q|k|v
  __shared__ __align__(16) bf16  sVt [C_ * VLD];        // [chan][token]
  __shared__ __align__(16) float sS  [NPAD * SLD];      // scores (one head)
  __shared__ __align__(16) bf16  sP  [NPAD * PLD];      // softmax probs
  __shared__ __align__(16) bf16  sO  [NPAD * OLD];      // attn out

  const int t    = threadIdx.x;
  const int lane = t & 31;
  const int wv   = t >> 5;
  const int hh   = (lane >> 4) & 1;
  const int ln16 = lane & 15;

  const int blk = blockIdx.x;
  const int b   = blk >> 10;            // 32*32 windows per image
  const int wy  = (blk >> 5) & 31;
  const int wx  = blk & 31;

  if (t == 0) { __builtin_prefetch(wqkv, 0, 2); __builtin_prefetch(wproj, 0, 2); }

  // ---- Stage transposed weights (bf16), float4 global loads ----
  {
    const float4* wq4 = (const float4*)wqkv;            // [96][288]
    for (int i4 = t; i4 < (3 * C_ * C_) / 4; i4 += 256) {
      const int flat = i4 * 4, k = flat / 288, n = flat % 288;
      float4 v = wq4[i4];
      sWq[(n + 0) * WLD + k] = (bf16)v.x;
      sWq[(n + 1) * WLD + k] = (bf16)v.y;
      sWq[(n + 2) * WLD + k] = (bf16)v.z;
      sWq[(n + 3) * WLD + k] = (bf16)v.w;
    }
    const float4* wp4 = (const float4*)wproj;           // [96][96]
    for (int i4 = t; i4 < (C_ * C_) / 4; i4 += 256) {
      const int flat = i4 * 4, k = flat / C_, n = flat % C_;
      float4 v = wp4[i4];
      sWp[(n + 0) * WLD + k] = (bf16)v.x;
      sWp[(n + 1) * WLD + k] = (bf16)v.y;
      sWp[(n + 2) * WLD + k] = (bf16)v.z;
      sWp[(n + 3) * WLD + k] = (bf16)v.w;
    }
  }

  // ---- LN1 + shifted gather: token m reads image (y'+3, x'+3) mod 224 ----
  if (t < NPAD) {
    if (t < NTOK) {
      const int row = (wy * 7 + t / 7 + SH_) % HW_;
      const int col = (wx * 7 + t % 7 + SH_) % HW_;
      const float4* px4 =
          (const float4*)(x + ((size_t)b * LTOT + (size_t)row * HW_ + col) * C_);
      float s = 0.f, s2 = 0.f;
      for (int c4 = 0; c4 < C_ / 4; ++c4) {
        float4 v = px4[c4];
        s  += v.x + v.y + v.z + v.w;
        s2 += v.x * v.x + v.y * v.y + v.z * v.z + v.w * v.w;
      }
      const float mu = s * (1.0f / C_);
      const float rs = rsqrtf(s2 * (1.0f / C_) - mu * mu + 1e-5f);
      for (int c4 = 0; c4 < C_ / 4; ++c4) {
        float4 v = px4[c4];
        const int c = c4 * 4;
        sX[t * XLD + c + 0] = (bf16)((v.x - mu) * rs * ln1g[c + 0] + ln1b[c + 0]);
        sX[t * XLD + c + 1] = (bf16)((v.y - mu) * rs * ln1g[c + 1] + ln1b[c + 1]);
        sX[t * XLD + c + 2] = (bf16)((v.z - mu) * rs * ln1g[c + 2] + ln1b[c + 2]);
        sX[t * XLD + c + 3] = (bf16)((v.w - mu) * rs * ln1g[c + 3] + ln1b[c + 3]);
      }
    } else {
      for (int c = 0; c < C_; ++c) sX[t * XLD + c] = (bf16)0.0f;
    }
  }
  __syncthreads();

  // ---- QKV GEMM: (64x96)@(96x288). 4x18 tiles, 9/wave, K=3; uniform store --
  for (int it = 0; it < 9; ++it) {
    const int tile = wv * 9 + it;
    const int mt = tile / 18, nt = tile % 18;
    v8f acc = {};
#pragma unroll
    for (int ks = 0; ks < 3; ++ks) {
      v16bf a  = load_frag(sX,  mt * 16 + ln16, XLD, ks * 32, lane);
      v16bf bm = load_frag(sWq, nt * 16 + ln16, WLD, ks * 32, lane);
      acc = wmma_bf16(a, bm, acc);
    }
    const int nc = nt * 16 + ln16;
    const float bias = bqkv[nc];
#pragma unroll
    for (int r = 0; r < 8; ++r)
      sQKV[(mt * 16 + r + 8 * hh) * QLD + nc] = (bf16)(acc[r] + bias);
  }
  __syncthreads();

  // ---- V transpose pass: sVt[chan][token] = sQKV[token][192+chan] ----
  for (int idx = t; idx < C_ * NPAD; idx += 256) {
    const int tok = idx & (NPAD - 1), ch = idx >> 6;
    sVt[ch * VLD + tok] = sQKV[tok * QLD + 2 * C_ + ch];
  }
  __syncthreads();

  // ---- Per-head attention ----
  const float scale = 0.17677669529663687f;  // 32^-0.5
  for (int hd = 0; hd < NH_; ++hd) {
    // S = Q_h @ K_h^T * scale : 4x4 tiles, 2 per wave, single K=32 WMMA
#pragma unroll
    for (int it = 0; it < 2; ++it) {
      const int tile = wv * 2 + it;
      const int mt = tile >> 2, nt = tile & 3;
      v16bf a  = load_frag(sQKV, mt * 16 + ln16, QLD, hd * HD_, lane);
      v16bf bm = load_frag(sQKV, nt * 16 + ln16, QLD, C_ + hd * HD_, lane);
      v8f acc = {};
      acc = wmma_bf16(a, bm, acc);
#pragma unroll
      for (int r = 0; r < 8; ++r)
        sS[(mt * 16 + r + 8 * hh) * SLD + nt * 16 + ln16] = acc[r] * scale;
    }
    __syncthreads();
    // Softmax over the 49 valid columns; one thread per row.
    if (t < NTOK) {
      float mx = -1e30f;
      for (int j = 0; j < NTOK; ++j) mx = fmaxf(mx, sS[t * SLD + j]);
      float sum = 0.f;
      for (int j = 0; j < NTOK; ++j) {
        float e = __expf(sS[t * SLD + j] - mx);
        sS[t * SLD + j] = e; sum += e;
      }
      const float inv = 1.0f / sum;
      for (int j = 0; j < NTOK; ++j) sP[t * PLD + j] = (bf16)(sS[t * SLD + j] * inv);
      for (int j = NTOK; j < NPAD; ++j) sP[t * PLD + j] = (bf16)0.0f;
    }
    __syncthreads();
    // O_h = P @ V_h : 4x2 tiles, 1 per wave, K=64 (2 steps)
    {
      const int mt = wv >> 1, nt = wv & 1;
      v8f acc = {};
#pragma unroll
      for (int ks = 0; ks < 2; ++ks) {
        v16bf a  = load_frag(sP,  mt * 16 + ln16, PLD, ks * 32, lane);
        v16bf bm = load_frag(sVt, hd * HD_ + nt * 16 + ln16, VLD, ks * 32, lane);
        acc = wmma_bf16(a, bm, acc);
      }
#pragma unroll
      for (int r = 0; r < 8; ++r)
        sO[(mt * 16 + r + 8 * hh) * OLD + hd * HD_ + nt * 16 + ln16] = (bf16)acc[r];
    }
    __syncthreads();
  }

  // ---- Projection: (64x96)@(96x96) + bias, scatter with reverse shift ----
  for (int it = 0; it < 3; ++it) {
    const int tile = wv * 3 + it;
    const int mt = tile / 6, nt = tile % 6;
    v8f acc = {};
#pragma unroll
    for (int ks = 0; ks < 3; ++ks) {
      v16bf a  = load_frag(sO,  mt * 16 + ln16, OLD, ks * 32, lane);
      v16bf bm = load_frag(sWp, nt * 16 + ln16, WLD, ks * 32, lane);
      acc = wmma_bf16(a, bm, acc);
    }
    const int nc = nt * 16 + ln16;
    const float bias = bproj[nc];
#pragma unroll
    for (int r = 0; r < 8; ++r) {
      const int m = mt * 16 + r + 8 * hh;
      if (m < NTOK) {
        const int row = (wy * 7 + m / 7 + SH_) % HW_;
        const int col = (wx * 7 + m % 7 + SH_) % HW_;
        hout[((size_t)b * LTOT + (size_t)row * HW_ + col) * C_ + nc] = acc[r] + bias;
      }
    }
  }
}

// ===========================================================================
// Kernel B: LN2 + MLP(GELU) + residuals. 64 tokens per WG, 256 threads.
// out = x + h + gelu(LN2(h) @ w1 + b1) @ w2 + b2   (in-place over h buffer)
// ===========================================================================
__global__ __launch_bounds__(256) void swin_mlp_kernel(
    const float* __restrict__ x, const float* __restrict__ hbuf,
    const float* __restrict__ ln2g, const float* __restrict__ ln2b,
    const float* __restrict__ w1, const float* __restrict__ b1,
    const float* __restrict__ w2, const float* __restrict__ b2,
    float* __restrict__ out) {
  __shared__ __align__(16) float sH [NPAD * C_];     // h rows, f32 (residual)
  __shared__ __align__(16) bf16  sY [NPAD * XLD];    // LN2(h), bf16
  __shared__ __align__(16) bf16  sW1[HID_ * WLD];    // w1^T: [n][k], k=96
  __shared__ __align__(16) bf16  sA [NPAD * ALD];    // gelu acts
  __shared__ __align__(16) bf16  sW2[C_ * ALD];      // w2^T: [n][k], k=384

  const int t    = threadIdx.x;
  const int lane = t & 31;
  const int wv   = t >> 5;
  const int hh   = (lane >> 4) & 1;
  const int ln16 = lane & 15;
  const size_t tok0 = (size_t)blockIdx.x * NPAD;

  if (t == 0) { __builtin_prefetch(w1, 0, 2); __builtin_prefetch(w2, 0, 2); }

  // ---- Stage h rows: contiguous global -> LDS (async DMA path on gfx1250) --
#if __has_builtin(__builtin_amdgcn_global_load_async_to_lds_b128)
  {
    const float* gsrc = hbuf + tok0 * C_;
    for (int i4 = t; i4 < (NPAD * C_) / 4; i4 += 256) {
      as1_v4i_p gp = (as1_v4i_p)(as1_void_p)(gsrc + i4 * 4);
      as3_v4i_p lp = (as3_v4i_p)(as3_void_p)(&sH[i4 * 4]);
      __builtin_amdgcn_global_load_async_to_lds_b128(gp, lp, 0, 0);
    }
    asm volatile("s_wait_asynccnt 0x0" ::: "memory");
  }
#else
  {
    const float4* h4 = (const float4*)(hbuf + tok0 * C_);
    float4* sH4 = (float4*)sH;
    for (int i4 = t; i4 < (NPAD * C_) / 4; i4 += 256) sH4[i4] = h4[i4];
  }
#endif

  // ---- Stage transposed MLP weights (bf16), float4 global loads ----
  {
    const float4* w14 = (const float4*)w1;             // [96][384]
    for (int i4 = t; i4 < (C_ * HID_) / 4; i4 += 256) {
      const int flat = i4 * 4, k = flat / HID_, n = flat % HID_;
      float4 v = w14[i4];
      sW1[(n + 0) * WLD + k] = (bf16)v.x;
      sW1[(n + 1) * WLD + k] = (bf16)v.y;
      sW1[(n + 2) * WLD + k] = (bf16)v.z;
      sW1[(n + 3) * WLD + k] = (bf16)v.w;
    }
    const float4* w24 = (const float4*)w2;             // [384][96]
    for (int i4 = t; i4 < (C_ * HID_) / 4; i4 += 256) {
      const int flat = i4 * 4, k = flat / C_, n = flat % C_;
      float4 v = w24[i4];
      sW2[(n + 0) * ALD + k] = (bf16)v.x;
      sW2[(n + 1) * ALD + k] = (bf16)v.y;
      sW2[(n + 2) * ALD + k] = (bf16)v.z;
      sW2[(n + 3) * ALD + k] = (bf16)v.w;
    }
  }
  __syncthreads();

  // ---- LN2 from LDS ----
  if (t < NPAD) {
    float s = 0.f, s2 = 0.f;
    for (int c = 0; c < C_; ++c) { float v = sH[t * C_ + c]; s += v; s2 += v * v; }
    const float mu = s * (1.0f / C_);
    const float rs = rsqrtf(s2 * (1.0f / C_) - mu * mu + 1e-5f);
    for (int c = 0; c < C_; ++c)
      sY[t * XLD + c] = (bf16)((sH[t * C_ + c] - mu) * rs * ln2g[c] + ln2b[c]);
  }
  __syncthreads();

  // GEMM1: (64x96)@(96x384), 4x24 tiles, 12/wave, K=3; exact GELU
  for (int it = 0; it < 12; ++it) {
    const int tile = wv * 12 + it;
    const int mt = tile / 24, nt = tile % 24;
    v8f acc = {};
#pragma unroll
    for (int ks = 0; ks < 3; ++ks) {
      v16bf a  = load_frag(sY,  mt * 16 + ln16, XLD, ks * 32, lane);
      v16bf bm = load_frag(sW1, nt * 16 + ln16, WLD, ks * 32, lane);
      acc = wmma_bf16(a, bm, acc);
    }
    const int nc = nt * 16 + ln16;
    const float bias = b1[nc];
#pragma unroll
    for (int r = 0; r < 8; ++r) {
      const int m = mt * 16 + r + 8 * hh;
      float v = acc[r] + bias;
      v = 0.5f * v * (1.0f + erff(v * 0.70710678118654752f));
      sA[m * ALD + nc] = (bf16)v;
    }
  }
  __syncthreads();

  // GEMM2: (64x384)@(384x96), 4x6 tiles, 3/wave, K=12 steps; + residuals
  for (int it = 0; it < 3; ++it) {
    const int tile = wv * 3 + it;
    const int mt = tile / 6, nt = tile % 6;
    v8f acc = {};
#pragma unroll
    for (int ks = 0; ks < 12; ++ks) {
      v16bf a  = load_frag(sA,  mt * 16 + ln16, ALD, ks * 32, lane);
      v16bf bm = load_frag(sW2, nt * 16 + ln16, ALD, ks * 32, lane);
      acc = wmma_bf16(a, bm, acc);
    }
    const int nc = nt * 16 + ln16;
    const float bias = b2[nc];
#pragma unroll
    for (int r = 0; r < 8; ++r) {
      const int m = mt * 16 + r + 8 * hh;
      out[(tok0 + m) * C_ + nc] =
          acc[r] + bias + sH[m * C_ + nc] + x[(tok0 + m) * C_ + nc];
    }
  }
}

// ===========================================================================
extern "C" void kernel_launch(void* const* d_in, const int* in_sizes, int n_in,
                              void* d_out, int out_size, void* d_ws, size_t ws_size,
                              hipStream_t stream) {
  (void)in_sizes; (void)n_in; (void)out_size; (void)d_ws; (void)ws_size;
  const float* x     = (const float*)d_in[0];
  // d_in[1]=H, d_in[2]=W, d_in[3]=shift_size (224/224/3, compiled in)
  const float* ln1g  = (const float*)d_in[4];
  const float* ln1b  = (const float*)d_in[5];
  const float* wqkv  = (const float*)d_in[6];
  const float* bqkv  = (const float*)d_in[7];
  const float* wproj = (const float*)d_in[8];
  const float* bproj = (const float*)d_in[9];
  const float* ln2g  = (const float*)d_in[10];
  const float* ln2b  = (const float*)d_in[11];
  const float* w1    = (const float*)d_in[12];
  const float* b1    = (const float*)d_in[13];
  const float* w2    = (const float*)d_in[14];
  const float* b2    = (const float*)d_in[15];
  float* out = (float*)d_out;

  // h (attention branch output) materialized into d_out, then consumed
  // in-place by the MLP kernel (each WG stages its own rows before writing).
  const int nwin = BB_ * 32 * 32;                 // 8192 windows
  swin_attn_kernel<<<nwin, 256, 0, stream>>>(x, ln1g, ln1b, wqkv, bqkv,
                                             wproj, bproj, out);
  const int nblk = (BB_ * LTOT) / NPAD;           // 6272 blocks of 64 tokens
  swin_mlp_kernel<<<nblk, 256, 0, stream>>>(x, out, ln2g, ln2b,
                                            w1, b1, w2, b2, out);
}
